// Model_52518860096471
// MI455X (gfx1250) — matrix operand order repaired and checked by reading, not compiled, not run on hardware
//
#include <hip/hip_runtime.h>
#include <hip/hip_bf16.h>

// Flash-attention forward for B=4,H=16,S=2048,D=64 (fp32 in/out).
// Matmuls on v_wmma_f32_16x16x32_bf16 (fp32 accumulate).
//
// grid = (B*H) * (S/64) blocks, 128 threads (4 waves, wave32).
// Each wave owns a 16-query tile; the block cooperatively stages each
// 32-key K/V chunk into LDS as bf16 (converted once, shared by 4 waves),
// double-buffered so chunk j+1's global loads overlap chunk j's compute.

typedef __attribute__((ext_vector_type(16))) __bf16 v16bf;
typedef __attribute__((ext_vector_type(8)))  float  v8f;

#define ATTN_S 2048
#define ATTN_D 64
#define ATTN_SCALE 3.0f
#define NCHUNK (ATTN_S / 32)

// XOR lane shuffle via ds_swizzle (group-of-32 mode: imm = xor<<10 | and_mask 0x1f).
// Masks 1/2/4/8 stay within 16-lane halves (wave32).
#define SWZ_XOR_F(v, mask) \
  __int_as_float(__builtin_amdgcn_ds_swizzle(__float_as_int(v), (((mask) << 10) | 0x1f)))

static __device__ __forceinline__ unsigned short bf16_bits(float f) {
  __bf16 b = (__bf16)f;
  return __builtin_bit_cast(unsigned short, b);
}
static __device__ __forceinline__ __bf16 bits_bf16(unsigned short u) {
  return __builtin_bit_cast(__bf16, u);
}
static __device__ __forceinline__ unsigned int pack_bf16x2(float lo, float hi) {
  return (unsigned int)bf16_bits(lo) | ((unsigned int)bf16_bits(hi) << 16);
}

// 32-byte fragment view: two uint4 LDS loads aliased to a v16bf operand.
union FragU {
  uint4        q[2];
  unsigned int u[8];
  v16bf        b;
};

__global__ __launch_bounds__(128)
void fa_fwd_kernel(const float* __restrict__ x1,
                   const float* __restrict__ x2,
                   float* __restrict__ out) {
  const int tid  = threadIdx.x;
  const int lane = tid & 31;
  const int wave = tid >> 5;
  const int m    = lane & 15;   // A-row / B-C-D-column index within a 16 tile
  const int half = lane >> 4;   // 16-lane half

  const int qt = blockIdx.x & 31;  // 32 query blocks of 64
  const int bh = blockIdx.x >> 5;  // B*H = 64
  const int q0 = qt * 64 + wave * 16;

  const float* Q = x1 + (size_t)bh * ATTN_S * ATTN_D;
  const float* K = x2 + (size_t)bh * ATTN_S * ATTN_D;
  const float* V = Q;  // reference reuses x1 as values
  float*       O = out + (size_t)bh * ATTN_S * ATTN_D;

  // Double-buffered bf16 chunk tiles (shared by all 4 waves):
  //  kb_lds[buf][key][d_pair]  : K chunk, row-major, 2 bf16 per uint (pad 36)
  //  vT_lds[buf][d][key]       : V chunk, transposed (pad 40) -> contiguous B-frag reads
  __shared__ __align__(16) unsigned int   kb_lds[2][32][36];
  __shared__ __align__(16) unsigned short vT_lds[2][64][40];
  // Per-wave P staging tile (16 x 32 bf16, padded stride 40).
  __shared__ __align__(16) unsigned short pbuf[4][16][40];

  // ---- Q A-fragments: two 16x32 bf16 tiles covering D=64 (loaded once) --
  // A 16x32 layout: lane holds row m; element pair (2t,2t+1) holds
  // K = half*16 + 2t + {0,1}.
  v16bf qa[2];
#pragma unroll
  for (int c = 0; c < 2; ++c) {
#pragma unroll
    for (int t = 0; t < 8; ++t) {
      const int k = (t << 1) + (half << 4);
      const float* p = Q + (size_t)(q0 + m) * ATTN_D + c * 32 + k;
      qa[c][2 * t]     = (__bf16)p[0];
      qa[c][2 * t + 1] = (__bf16)p[1];
    }
  }

  v8f acc[4] = {};        // output accumulators: 4 N-tiles of 16 (D=64)
  float mrow[8], lrow[8]; // per-row running max / sum (lane-aligned with C layout)
#pragma unroll
  for (int v = 0; v < 8; ++v) { mrow[v] = -1e30f; lrow[v] = 0.0f; }

  // ---- software pipeline: preload chunk 0 into registers -----------------
  float4 kreg[4], vreg[4];
#pragma unroll
  for (int i = 0; i < 4; ++i) {
    const int idx = i * 128 + tid;        // 512 float4 per chunk
    const int row = idx >> 4;             // key row 0..31
    const int c4  = idx & 15;             // float4 column 0..15
    kreg[i] = *(const float4*)(K + (size_t)row * ATTN_D + c4 * 4);
    vreg[i] = *(const float4*)(V + (size_t)row * ATTN_D + c4 * 4);
  }

  for (int j = 0; j < NCHUNK; ++j) {
    const int buf  = j & 1;
    const int key0 = j * 32;

    // ---- stage chunk j (registers -> bf16 LDS), conversion shared -------
#pragma unroll
    for (int i = 0; i < 4; ++i) {
      const int idx = i * 128 + tid;
      const int row = idx >> 4;
      const int c4  = idx & 15;
      kb_lds[buf][row][c4 * 2]     = pack_bf16x2(kreg[i].x, kreg[i].y);
      kb_lds[buf][row][c4 * 2 + 1] = pack_bf16x2(kreg[i].z, kreg[i].w);
      vT_lds[buf][c4 * 4 + 0][row] = bf16_bits(vreg[i].x);
      vT_lds[buf][c4 * 4 + 1][row] = bf16_bits(vreg[i].y);
      vT_lds[buf][c4 * 4 + 2][row] = bf16_bits(vreg[i].z);
      vT_lds[buf][c4 * 4 + 3][row] = bf16_bits(vreg[i].w);
    }
    __syncthreads();  // stores visible; also separates compute(j-1) from store(j+1)

    // ---- issue global loads for chunk j+1 (latency hidden by compute) ---
    if (j + 1 < NCHUNK) {
      const int kn = (j + 1) * 32;
#pragma unroll
      for (int i = 0; i < 4; ++i) {
        const int idx = i * 128 + tid;
        const int row = idx >> 4;
        const int c4  = idx & 15;
        kreg[i] = *(const float4*)(K + (size_t)(kn + row) * ATTN_D + c4 * 4);
        vreg[i] = *(const float4*)(V + (size_t)(kn + row) * ATTN_D + c4 * 4);
      }
    }
    // Warm L2 for chunk j+2 (speculative prefetch; global_prefetch_b8).
    if (j + 2 < NCHUNK) {
      const int kp = (j + 2) * 32;
      __builtin_prefetch(K + (size_t)(kp + (tid >> 2)) * ATTN_D, 0, 1);
      __builtin_prefetch(V + (size_t)(kp + (tid >> 2)) * ATTN_D, 0, 1);
    }

    // ---- scores: two 16x16 f32 tiles = Q (16xD) @ K_chunk^T -------------
    v8f s[2];
#pragma unroll
    for (int nk = 0; nk < 2; ++nk) {
      v8f acc_s = {};
#pragma unroll
      for (int c = 0; c < 2; ++c) {
        // B 32x16: lane col n=m (key nk*16+m), element e = d c*32+half*16+e
        // -> 8 consecutive uints in kb_lds -> two b128 LDS loads.
        FragU kf;
        const unsigned int* p = &kb_lds[buf][nk * 16 + m][(c << 4) + (half << 3)];
        kf.q[0] = *(const uint4*)(p);
        kf.q[1] = *(const uint4*)(p + 4);
        acc_s = __builtin_amdgcn_wmma_f32_16x16x32_bf16(
            false, qa[c], false, kf.b, (short)0, acc_s, false, false);
      }
      s[nk] = acc_s;
    }

    // ---- online softmax (row-aligned per lane) --------------------------
#pragma unroll
    for (int v = 0; v < 8; ++v) {
      float t0 = s[0][v] * ATTN_SCALE;
      float t1 = s[1][v] * ATTN_SCALE;

      float mx = fmaxf(t0, t1);
      mx = fmaxf(mx, SWZ_XOR_F(mx, 1));
      mx = fmaxf(mx, SWZ_XOR_F(mx, 2));
      mx = fmaxf(mx, SWZ_XOR_F(mx, 4));
      mx = fmaxf(mx, SWZ_XOR_F(mx, 8));

      const float mn    = fmaxf(mrow[v], mx);
      const float alpha = __expf(mrow[v] - mn);
      mrow[v] = mn;

      const float p0 = __expf(t0 - mn);
      const float p1 = __expf(t1 - mn);

      float rs = p0 + p1;
      rs += SWZ_XOR_F(rs, 1);
      rs += SWZ_XOR_F(rs, 2);
      rs += SWZ_XOR_F(rs, 4);
      rs += SWZ_XOR_F(rs, 8);
      lrow[v] = lrow[v] * alpha + rs;

#pragma unroll
      for (int nt = 0; nt < 4; ++nt) acc[nt][v] *= alpha;

      // Stage P as bf16: C layout row = v + half*8, cols nk*16 + m.
      const int row = v + half * 8;
      pbuf[wave][row][m]      = bf16_bits(p0);
      pbuf[wave][row][16 + m] = bf16_bits(p1);
    }

    // Same-wave cross-lane LDS write -> read ordering (per-wave tile).
    asm volatile("s_wait_dscnt 0" ::: "memory");

    // ---- read P back as A-fragment (16x32 bf16) -------------------------
    v16bf pa;
#pragma unroll
    for (int t = 0; t < 8; ++t) {
      const int k = (t << 1) + (half << 4);
      pa[2 * t]     = bits_bf16(pbuf[wave][m][k]);
      pa[2 * t + 1] = bits_bf16(pbuf[wave][m][k + 1]);
    }

    // ---- acc += P (16x32) @ V_chunk (32x64), 4 N-tiles ------------------
#pragma unroll
    for (int nt = 0; nt < 4; ++nt) {
      // B 32x16 from transposed V: element e = key half*16+e, col d = nt*16+m
      // -> 16 consecutive bf16 -> two b128 LDS loads.
      FragU vf;
      const unsigned short* vp = &vT_lds[buf][nt * 16 + m][half << 4];
      vf.q[0] = *(const uint4*)(vp);
      vf.q[1] = *(const uint4*)(vp + 8);
      acc[nt] = __builtin_amdgcn_wmma_f32_16x16x32_bf16(
          false, pa, false, vf.b, (short)0, acc[nt], false, false);
    }
  }

  // ---- normalize and store (C layout: VGPR v -> row v + half*8) ---------
#pragma unroll
  for (int v = 0; v < 8; ++v) {
    const float inv = 1.0f / lrow[v];
    const int row = v + half * 8;
#pragma unroll
    for (int nt = 0; nt < 4; ++nt)
      O[(size_t)(q0 + row) * ATTN_D + nt * 16 + m] = acc[nt][v] * inv;
  }
}

extern "C" void kernel_launch(void* const* d_in, const int* in_sizes, int n_in,
                              void* d_out, int out_size, void* d_ws, size_t ws_size,
                              hipStream_t stream) {
  const float* x1 = (const float*)d_in[0];
  const float* x2 = (const float*)d_in[1];
  float* out = (float*)d_out;

  // B*H = 64 slices, 32 query-blocks of 64 each -> 2048 blocks of 128 threads.
  dim3 grid(64 * 32);
  dim3 block(128);
  fa_fwd_kernel<<<grid, block, 0, stream>>>(x1, x2, out);
}
